// MultiHeadAttention_21303037788208
// MI455X (gfx1250) — matrix operand order repaired and checked by reading, not compiled, not run on hardware
//
#include <hip/hip_runtime.h>

#define BB 2
#define SS 2048
#define DM 1024
#define NH 16
#define DKV 64

typedef __attribute__((ext_vector_type(16))) __bf16 v16bf;
typedef __attribute__((ext_vector_type(8)))  float  v8f;
typedef __attribute__((ext_vector_type(8)))  unsigned int v8u;
typedef int v4i_ __attribute__((vector_size(16)));
typedef __attribute__((address_space(1))) v4i_ as1_v4i;
typedef __attribute__((address_space(3))) v4i_ as3_v4i;

__device__ __forceinline__ unsigned short f2bf(float f) {
  // native f32 -> bf16 RNE conversion (v_cvt on gfx1250)
  __bf16 h = (__bf16)f;
  return __builtin_bit_cast(unsigned short, h);
}

__device__ __forceinline__ v8f vzero8() {
  v8f r;
#pragma unroll
  for (int i = 0; i < 8; ++i) r[i] = 0.0f;
  return r;
}

// Load a 32-byte bf16 fragment (v16bf) as two 16B chunks.
__device__ __forceinline__ v16bf frag2(const unsigned short* p0, const unsigned short* p1) {
  uint4 a = *(const uint4*)p0;
  uint4 b = *(const uint4*)p1;
  v8u r;
  r[0] = a.x; r[1] = a.y; r[2] = a.z; r[3] = a.w;
  r[4] = b.x; r[5] = b.y; r[6] = b.z; r[7] = b.w;
  return __builtin_bit_cast(v16bf, r);
}

__device__ __forceinline__ v8f wmma_bf16(v16bf a, v16bf b, v8f c) {
  return __builtin_amdgcn_wmma_f32_16x16x32_bf16(false, a, false, b, (short)0, c,
                                                 false, false);
}

// Async global -> LDS copy (16B per lane), tracked by ASYNCcnt.
__device__ __forceinline__ void async_g2l_b128(const unsigned short* g,
                                               unsigned short* l) {
#if __has_builtin(__builtin_amdgcn_global_load_async_to_lds_b128)
  __builtin_amdgcn_global_load_async_to_lds_b128(
      (as1_v4i*)(uintptr_t)g,            // inttoptr into AS1 (global)
      (as3_v4i*)(void*)l,                // addrspacecast generic -> AS3 (LDS)
      0, 0);
#else
  unsigned int loff = (unsigned int)(unsigned long long)
      (__attribute__((address_space(3))) unsigned short*)l;
  unsigned long long ga = (unsigned long long)g;
  asm volatile("global_load_async_to_lds_b128 %0, %1, off"
               :: "v"(loff), "v"(ga) : "memory");
#endif
}

__device__ __forceinline__ void wait_async0() {
#if __has_builtin(__builtin_amdgcn_s_wait_asynccnt)
  __builtin_amdgcn_s_wait_asynccnt(0);
#else
  asm volatile("s_wait_asynccnt 0x0" ::: "memory");
#endif
}

// Reductions across the 16 lanes owning one C-layout row (xor masks 1,2,4,8
// never cross the lane-half boundary on wave32).
__device__ __forceinline__ float rowmax16(float v) {
#pragma unroll
  for (int m = 1; m < 16; m <<= 1) v = fmaxf(v, __shfl_xor(v, m, 32));
  return v;
}
__device__ __forceinline__ float rowsum16(float v) {
#pragma unroll
  for (int m = 1; m < 16; m <<= 1) v += __shfl_xor(v, m, 32);
  return v;
}

// ---------------------------------------------------------------------------
// Kernel 1: QKV projection.  out(t,h) tile = X[128 x 1024] * W_t,h[1024 x 64]
// bf16 WMMA with f32 accumulation.  Q,K stored row-major [bh][s][dk] bf16.
// V stored transposed [bh][dv][s] bf16 so attention's V slabs are contiguous.
// ---------------------------------------------------------------------------
__global__ __launch_bounds__(256) void proj_qkv_kernel(
    const float* __restrict__ x, const float* __restrict__ Wq,
    const float* __restrict__ Wk, const float* __restrict__ Wv,
    unsigned short* __restrict__ Qh, unsigned short* __restrict__ Kh,
    unsigned short* __restrict__ Vt) {
  constexpr int XSTR = 40;   // row stride (bf16 elems) for X tile, 16B aligned
  constexpr int WSTR = 40;   // col stride (bf16 elems) for W tile, 16B aligned
  __shared__ unsigned short Xl[128 * XSTR];
  __shared__ unsigned short Wl[64 * WSTR];

  const int tid  = threadIdx.x;
  const int wave = tid >> 5;
  const int lane = tid & 31;
  const int hf   = lane >> 4;
  const int ln   = lane & 15;
  const int m0   = blockIdx.x * 128;             // row tile in flattened [B*S]
  const int t    = blockIdx.y / NH;              // 0=Q 1=K 2=V
  const int h    = blockIdx.y % NH;
  const float* W = (t == 0 ? Wq : (t == 1 ? Wk : Wv)) + (size_t)h * DM * DKV;

  v8f c0 = vzero8(), c1 = vzero8(), c2 = vzero8(), c3 = vzero8();

  for (int k0 = 0; k0 < DM; k0 += 32) {
    { // stage X tile (128 x 32 f32 -> bf16, row-major)
      const int row  = tid >> 1;
      const int cofs = (tid & 1) * 16;
      const float* xp = x + (size_t)(m0 + row) * DM + k0 + cofs;
      float4 f0 = ((const float4*)xp)[0];
      float4 f1 = ((const float4*)xp)[1];
      float4 f2 = ((const float4*)xp)[2];
      float4 f3 = ((const float4*)xp)[3];
      alignas(16) unsigned short tmp[16];
      tmp[0]  = f2bf(f0.x); tmp[1]  = f2bf(f0.y); tmp[2]  = f2bf(f0.z); tmp[3]  = f2bf(f0.w);
      tmp[4]  = f2bf(f1.x); tmp[5]  = f2bf(f1.y); tmp[6]  = f2bf(f1.z); tmp[7]  = f2bf(f1.w);
      tmp[8]  = f2bf(f2.x); tmp[9]  = f2bf(f2.y); tmp[10] = f2bf(f2.z); tmp[11] = f2bf(f2.w);
      tmp[12] = f2bf(f3.x); tmp[13] = f2bf(f3.y); tmp[14] = f2bf(f3.z); tmp[15] = f2bf(f3.w);
      uint4* d = (uint4*)&Xl[row * XSTR + cofs];
      d[0] = ((const uint4*)tmp)[0];
      d[1] = ((const uint4*)tmp)[1];
    }
    { // stage W tile (32 x 64 f32 -> bf16, column-major for B-frags)
      const int dd = tid >> 3;         // k within tile 0..31
      const int c8 = (tid & 7) * 8;    // col 0..56
      const float* wp = W + (size_t)(k0 + dd) * DKV + c8;
      float4 g0 = ((const float4*)wp)[0];
      float4 g1 = ((const float4*)wp)[1];
      unsigned short wb[8];
      wb[0] = f2bf(g0.x); wb[1] = f2bf(g0.y); wb[2] = f2bf(g0.z); wb[3] = f2bf(g0.w);
      wb[4] = f2bf(g1.x); wb[5] = f2bf(g1.y); wb[6] = f2bf(g1.z); wb[7] = f2bf(g1.w);
#pragma unroll
      for (int i = 0; i < 8; ++i) Wl[(c8 + i) * WSTR + dd] = wb[i];
    }
    __syncthreads();
    // A-frag: 16 rows (this wave) x 32 k
    const unsigned short* ap = &Xl[(wave * 16 + ln) * XSTR + hf * 8];
    v16bf a = frag2(ap, ap + 16);
    // 4 B-frags: 32 k x 16 n each
    const unsigned short* b0 = &Wl[(0 * 16 + ln) * WSTR + hf * 8];
    const unsigned short* b1 = &Wl[(1 * 16 + ln) * WSTR + hf * 8];
    const unsigned short* b2 = &Wl[(2 * 16 + ln) * WSTR + hf * 8];
    const unsigned short* b3 = &Wl[(3 * 16 + ln) * WSTR + hf * 8];
    c0 = wmma_bf16(a, frag2(b0, b0 + 16), c0);
    c1 = wmma_bf16(a, frag2(b1, b1 + 16), c1);
    c2 = wmma_bf16(a, frag2(b2, b2 + 16), c2);
    c3 = wmma_bf16(a, frag2(b3, b3 + 16), c3);
    __syncthreads();
  }

  const int b  = m0 / SS;
  const int s0 = (m0 % SS) + wave * 16;
  v8f cc[4] = {c0, c1, c2, c3};
  if (t < 2) {
    unsigned short* dst = (t == 0 ? Qh : Kh) + (size_t)(b * NH + h) * SS * DKV;
#pragma unroll
    for (int tt = 0; tt < 4; ++tt)
#pragma unroll
      for (int r = 0; r < 8; ++r)
        dst[(size_t)(s0 + r + 8 * hf) * DKV + tt * 16 + ln] = f2bf(cc[tt][r]);
  } else {
    unsigned short* dst = Vt + (size_t)(b * NH + h) * DKV * SS;
#pragma unroll
    for (int tt = 0; tt < 4; ++tt)
#pragma unroll
      for (int r = 0; r < 8; ++r)
        dst[(size_t)(tt * 16 + ln) * SS + (s0 + r + 8 * hf)] = f2bf(cc[tt][r]);
  }
}

// ---------------------------------------------------------------------------
// Kernel 2: flash attention (no 1/sqrt(dk) scale, per reference).
// Workgroup = 8 waves; each wave owns 16 query rows.  The 32-key K/V slabs are
// staged into LDS once per workgroup with async global->LDS copies (ASYNCcnt),
// double-buffered so the copy of slab j+32 overlaps the WMMAs on slab j.
// Per iter per wave: 4 WMMAs for Q.K^T, 4 WMMAs for P.V.
// ---------------------------------------------------------------------------
__global__ __launch_bounds__(256) void flash_attn_kernel(
    const unsigned short* __restrict__ Qh, const unsigned short* __restrict__ Kh,
    const unsigned short* __restrict__ Vt, float* __restrict__ out) {
  constexpr int KSTR = 72;   // K slab row stride (bf16) : 32 keys x 64 dk
  constexpr int VSTR = 40;   // V slab row stride (bf16) : 64 dv  x 32 keys
  __shared__ unsigned short KL[2][32 * KSTR];
  __shared__ unsigned short VL[2][64 * VSTR];
  __shared__ unsigned short PL[8 * 16 * 32];     // per-wave 16x32 bf16 P tile

  const int tid  = threadIdx.x;
  const int wave = tid >> 5;
  const int lane = tid & 31;
  const int hf   = lane >> 4;
  const int ln   = lane & 15;
  const int bh   = blockIdx.y;            // b*NH + h
  const int b    = bh >> 4;
  const int h    = bh & 15;
  const int q0   = blockIdx.x * 128 + wave * 16;

  const unsigned short* Qp = Qh + ((size_t)bh * SS + q0) * DKV;
  const unsigned short* Kp = Kh + (size_t)bh * SS * DKV;
  const unsigned short* Vp = Vt + (size_t)bh * DKV * SS;
  unsigned short* myP = &PL[wave * (16 * 32)];

  // per-thread staging coordinates (whole workgroup moves one 4KB slab each)
  const int skey = tid >> 3, skc = (tid & 7) * 8;   // K: 32 rows x 8 chunks
  const int sdv  = tid >> 2, svc = (tid & 3) * 8;   // V: 64 rows x 4 chunks

  auto stage = [&](int j, int buf) {
    async_g2l_b128(Kp + (size_t)(j + skey) * DKV + skc, &KL[buf][skey * KSTR + skc]);
    async_g2l_b128(Vp + (size_t)sdv * SS + j + svc,     &VL[buf][sdv  * VSTR + svc]);
  };

  // Q A-frags (dk 0..31 and 32..63) kept resident in VGPRs
  const unsigned short* qp0 = Qp + (size_t)ln * DKV + hf * 8;
  const v16bf qa0 = frag2(qp0, qp0 + 16);
  const v16bf qa1 = frag2(qp0 + 32, qp0 + 48);

  v8f o0 = vzero8(), o1 = vzero8(), o2 = vzero8(), o3 = vzero8();
  const v8f zacc = vzero8();
  float mrow[8], lrow[8];
#pragma unroll
  for (int r = 0; r < 8; ++r) { mrow[r] = -1.0e30f; lrow[r] = 0.0f; }

  stage(0, 0);
  int pb = 0;
  for (int j = 0; j < SS; j += 32, pb ^= 1) {
    wait_async0();        // own slab-portion landed in LDS
    __syncthreads();      // everyone's portion landed; prior compute finished
    if (j + 32 < SS) stage(j + 32, pb ^ 1);   // overlap next copy with compute

    // ---- scores: S = Q (16x64) . K^T (64x16) for two 16-key blocks ----
    const unsigned short* k00 = &KL[pb][(0 * 16 + ln) * KSTR + 0  + hf * 8];
    const unsigned short* k01 = &KL[pb][(0 * 16 + ln) * KSTR + 32 + hf * 8];
    const unsigned short* k10 = &KL[pb][(1 * 16 + ln) * KSTR + 0  + hf * 8];
    const unsigned short* k11 = &KL[pb][(1 * 16 + ln) * KSTR + 32 + hf * 8];
    v8f s1 = wmma_bf16(qa0, frag2(k00, k00 + 16), zacc);
    s1     = wmma_bf16(qa1, frag2(k01, k01 + 16), s1);
    v8f s2 = wmma_bf16(qa0, frag2(k10, k10 + 16), zacc);
    s2     = wmma_bf16(qa1, frag2(k11, k11 + 16), s2);

    // ---- online softmax over this 32-key slab ----
    float nm[8], sc[8], p1v[8], p2v[8];
#pragma unroll
    for (int r = 0; r < 8; ++r) {
      float tmax = rowmax16(fmaxf(s1[r], s2[r]));
      nm[r] = fmaxf(mrow[r], tmax);
      sc[r] = __expf(mrow[r] - nm[r]);
      mrow[r] = nm[r];
      p1v[r] = __expf(s1[r] - nm[r]);
      p2v[r] = __expf(s2[r] - nm[r]);
      lrow[r] = lrow[r] * sc[r] + rowsum16(p1v[r] + p2v[r]);
    }
    v8f scv;
#pragma unroll
    for (int r = 0; r < 8; ++r) scv[r] = sc[r];
    o0 *= scv; o1 *= scv; o2 *= scv; o3 *= scv;

    // ---- transpose P: C-layout -> A-frag layout via per-wave LDS ----
#pragma unroll
    for (int r = 0; r < 8; ++r) {
      const int m = r + 8 * hf;
      myP[m * 32 + ln]      = f2bf(p1v[r]);
      myP[m * 32 + 16 + ln] = f2bf(p2v[r]);
    }
    asm volatile("s_wait_dscnt 0" ::: "memory");   // same-wave DS ordering
    const unsigned short* pp0 = myP + ln * 32 + hf * 8;
    v16bf pa = frag2(pp0, pp0 + 16);
    asm volatile("" ::: "memory");

    // ---- O += P (16x32) . V (32x64) from the LDS V slab ----
    const unsigned short* v0p = &VL[pb][(0 * 16 + ln) * VSTR + hf * 8];
    const unsigned short* v1p = &VL[pb][(1 * 16 + ln) * VSTR + hf * 8];
    const unsigned short* v2p = &VL[pb][(2 * 16 + ln) * VSTR + hf * 8];
    const unsigned short* v3p = &VL[pb][(3 * 16 + ln) * VSTR + hf * 8];
    o0 = wmma_bf16(pa, frag2(v0p, v0p + 16), o0);
    o1 = wmma_bf16(pa, frag2(v1p, v1p + 16), o1);
    o2 = wmma_bf16(pa, frag2(v2p, v2p + 16), o2);
    o3 = wmma_bf16(pa, frag2(v3p, v3p + 16), o3);
  }

  // ---- epilogue: normalize and write fp32 output [B,S,H*DV] ----
  float invl[8];
#pragma unroll
  for (int r = 0; r < 8; ++r) invl[r] = 1.0f / lrow[r];
  float* op = out + ((size_t)b * SS + q0) * DM + (size_t)h * DKV;
  v8f oo[4] = {o0, o1, o2, o3};
#pragma unroll
  for (int tt = 0; tt < 4; ++tt)
#pragma unroll
    for (int r = 0; r < 8; ++r)
      op[(size_t)(r + 8 * hf) * DM + tt * 16 + ln] = oo[tt][r] * invl[r];
}

// ---------------------------------------------------------------------------
extern "C" void kernel_launch(void* const* d_in, const int* in_sizes, int n_in,
                              void* d_out, int out_size, void* d_ws, size_t ws_size,
                              hipStream_t stream) {
  const float* x  = (const float*)d_in[0];
  const float* Wq = (const float*)d_in[1];
  const float* Wk = (const float*)d_in[2];
  const float* Wv = (const float*)d_in[3];
  float* out = (float*)d_out;

  const size_t per = (size_t)BB * NH * SS * DKV;      // elements per buffer
  unsigned short* Qh = (unsigned short*)d_ws;
  unsigned short* Kh = Qh + per;
  unsigned short* Vt = Kh + per;                      // V stored [bh][dv][s]

  proj_qkv_kernel<<<dim3((BB * SS) / 128, 3 * NH), 256, 0, stream>>>(
      x, Wq, Wk, Wv, Qh, Kh, Vt);
  flash_attn_kernel<<<dim3(SS / 128, BB * NH), 256, 0, stream>>>(
      Qh, Kh, Vt, out);
}
